// LM_6347961664088
// MI455X (gfx1250) — matrix operand order repaired and checked by reading, not compiled, not run on hardware
//
#include <hip/hip_runtime.h>
#include <hip/hip_bf16.h>
#include <math.h>

// ---------------- types ----------------
typedef _Float16 v16h __attribute__((ext_vector_type(16)));
typedef _Float16 v8h  __attribute__((ext_vector_type(8)));
typedef _Float16 v4h  __attribute__((ext_vector_type(4)));
typedef float    v8f  __attribute__((ext_vector_type(8)));

#define WMMA_F16(a, b, c) \
  __builtin_amdgcn_wmma_f32_16x16x32_f16(false, (a), false, (b), (short)0, (c), false, false)

// ---------------- model constants ----------------
static constexpr int Vv   = 128;
static constexpr int Ee   = 1024;
static constexpr int Ll   = 8;
static constexpr int Hh   = 16;
static constexpr int Tt   = 2048;
static constexpr int Bb   = 2;
static constexpr int NTOK = Bb * Tt;   // 4096
static constexpr int E3   = 3 * Ee;    // 3072

// ---------------- CDNA5 async copy helpers ----------------
// LDS-aperture flat addresses carry the wave-relative LDS byte offset in
// their low 32 bits (ISA 10.2), so truncation yields the DS address the
// async-copy instruction wants.
__device__ __forceinline__ void async_copy_b128(const _Float16* gsrc, _Float16* lds_dst)
{
    uint32_t l = (uint32_t)(uintptr_t)lds_dst;
    uint64_t g = (uint64_t)(uintptr_t)gsrc;
    asm volatile("global_load_async_to_lds_b128 %0, %1, off"
                 :: "v"(l), "v"(g) : "memory");
}

__device__ __forceinline__ void wait_async0()
{
#if __has_builtin(__builtin_amdgcn_s_wait_asynccnt)
    __builtin_amdgcn_s_wait_asynccnt(0);
#else
    asm volatile("s_wait_asynccnt 0x0" ::: "memory");
#endif
}

// Allow one outstanding async op (the one just issued for the *next* tile):
// async loads complete in order, so <=1 means the previous tile has landed.
__device__ __forceinline__ void wait_async1()
{
#if __has_builtin(__builtin_amdgcn_s_wait_asynccnt)
    __builtin_amdgcn_s_wait_asynccnt(1);
#else
    asm volatile("s_wait_asynccnt 0x1" ::: "memory");
#endif
}

// ---------------- GEMM: C[M,N] = act(A[M,K] @ W[K,N] + bias), all-f16 data ----------------
// Double-buffered LDS, software pipelined: the async DMA for tile i+1 runs
// underneath the WMMAs for tile i.
#define BM 64
#define BN 128
#define BK 32

template <int OUT16>
__global__ __launch_bounds__(256)
void gemm_f16_kernel(const _Float16* __restrict__ A, const _Float16* __restrict__ W,
                     const float* __restrict__ bias, void* __restrict__ Cout,
                     int M, int N, int K, int relu)
{
    __shared__ __align__(16) _Float16 As[2][BM][BK];   // row-major [m][k]
    __shared__ __align__(16) _Float16 Bs[2][BN][BK];   // transposed [n][k]

    const int tid      = threadIdx.x;
    const int lane     = tid & 31;
    const int wave     = tid >> 5;
    const int waveM    = wave >> 2;     // 0..1
    const int waveN    = wave & 3;      // 0..3
    const int laneMod  = lane & 15;
    const int laneHalf = lane >> 4;

    const int rowBase = blockIdx.y * BM;
    const int colBase = blockIdx.x * BN;

    // staging coordinates
    const int ar  = tid >> 2;            // 0..63   A row
    const int ac8 = (tid & 3) * 8;       // 0,8,16,24 (half index, 16B chunk)
    const int bcn = tid >> 1;            // 0..127  B column
    const int bkh = (tid & 1) * 16;      // 0 or 16 (k-half)

    v8f acc[2][2] = {};

    const int nk = K / BK;

    // prologue: stage tile 0 into buffer 0
    {
        async_copy_b128(A + (size_t)(rowBase + ar) * K + ac8, &As[0][ar][ac8]);
        const _Float16* src = W + (size_t)bkh * N + colBase + bcn;
        __align__(16) _Float16 tmp[16];
        #pragma unroll
        for (int j = 0; j < 16; ++j) tmp[j] = src[(size_t)j * N];
        *(v8h*)&Bs[0][bcn][bkh]     = *(const v8h*)&tmp[0];
        *(v8h*)&Bs[0][bcn][bkh + 8] = *(const v8h*)&tmp[8];
    }

    for (int i = 0; i < nk; ++i) {
        const int cur  = i & 1;
        const bool more = (i + 1 < nk);

        // ---- stage tile i+1 into the other buffer (overlaps tile-i compute) ----
        if (more) {
            const int k1 = (i + 1) * BK;
            async_copy_b128(A + (size_t)(rowBase + ar) * K + k1 + ac8,
                            &As[cur ^ 1][ar][ac8]);
            const _Float16* src = W + (size_t)(k1 + bkh) * N + colBase + bcn;
            __align__(16) _Float16 tmp[16];
            #pragma unroll
            for (int j = 0; j < 16; ++j) tmp[j] = src[(size_t)j * N];
            *(v8h*)&Bs[cur ^ 1][bcn][bkh]     = *(const v8h*)&tmp[0];
            *(v8h*)&Bs[cur ^ 1][bcn][bkh + 8] = *(const v8h*)&tmp[8];
            if (i + 2 < nk) __builtin_prefetch(src + (size_t)BK * N, 0, 1);
        }

        // tile i's async copy (issued last iteration / prologue) has landed once
        // at most the tile-(i+1) copy is still outstanding.
        if (more) wait_async1(); else wait_async0();
        __syncthreads();

        // ---- fragments (all ds_load_b128) ----
        v16h aF[2], bF[2];
        #pragma unroll
        for (int mt = 0; mt < 2; ++mt) {
            int r = waveM * 32 + mt * 16 + laneMod;
            v8h lo = *(const v8h*)&As[cur][r][laneHalf * 8];
            v8h hi = *(const v8h*)&As[cur][r][16 + laneHalf * 8];
            #pragma unroll
            for (int j = 0; j < 8; ++j) { aF[mt][j] = lo[j]; aF[mt][j + 8] = hi[j]; }
        }
        #pragma unroll
        for (int nt = 0; nt < 2; ++nt) {
            int cn = waveN * 32 + nt * 16 + laneMod;
            v8h lo = *(const v8h*)&Bs[cur][cn][laneHalf * 8];
            v8h hi = *(const v8h*)&Bs[cur][cn][16 + laneHalf * 8];
            #pragma unroll
            for (int j = 0; j < 8; ++j) { bF[nt][j] = lo[j]; bF[nt][j + 8] = hi[j]; }
        }
        #pragma unroll
        for (int mt = 0; mt < 2; ++mt)
            #pragma unroll
            for (int nt = 0; nt < 2; ++nt)
                acc[mt][nt] = WMMA_F16(aF[mt], bF[nt], acc[mt][nt]);

        __syncthreads();   // safe to overwrite buf[cur] next iteration
    }

    // ---- epilogue: bias (+ optional relu), f16 or f32 store ----
    #pragma unroll
    for (int mt = 0; mt < 2; ++mt) {
        #pragma unroll
        for (int nt = 0; nt < 2; ++nt) {
            int col = colBase + waveN * 32 + nt * 16 + laneMod;
            float bv = bias[col];
            #pragma unroll
            for (int i = 0; i < 8; ++i) {
                int row = rowBase + waveM * 32 + mt * 16 + laneHalf * 8 + i;
                float v = acc[mt][nt][i] + bv;
                if (relu) v = fmaxf(v, 0.0f);
                if (OUT16) ((_Float16*)Cout)[(size_t)row * N + col] = (_Float16)v;
                else       ((float*)Cout)[(size_t)row * N + col] = v;
            }
        }
    }
}

// ---------------- attention (flash-style, faithful tril-multiply mask) ----------------
// qkvh layout: [B, T, 3E], per token row head h occupies [h*192, h*192+192): q|k|v 64 each.
// One wave handles a 16-row q tile; loops over keys in chunks of 32. Masked scores become
// exactly 0 and still participate in the softmax (reference multiplies scores by tril).
// The V chunk is async-DMA'd into LDS at the top of each iteration so the copy overlaps
// the score WMMAs + softmax VALU work.
__global__ __launch_bounds__(32)
void attn_kernel(const _Float16* __restrict__ qkvh, _Float16* __restrict__ o, float scale)
{
    __shared__ __align__(16) _Float16 Pt[16][32];
    __shared__ __align__(16) _Float16 Vs[32][64];

    const int lane     = threadIdx.x;
    const int laneMod  = lane & 15;
    const int laneHalf = lane >> 4;
    const int qBase    = blockIdx.x * 16;
    const int h        = blockIdx.y;
    const int b        = blockIdx.z;
    const size_t rowStride = (size_t)E3;
    const size_t headOff   = (size_t)h * 192;

    // Q A-fragments (hd 0..31 and 32..63)
    v16h aQ0, aQ1;
    {
        const _Float16* qp = qkvh + ((size_t)(b * Tt + qBase + laneMod)) * rowStride + headOff;
        #pragma unroll
        for (int j = 0; j < 8; ++j) {
            aQ0[j]     = qp[laneHalf * 8 + j];
            aQ0[j + 8] = qp[16 + laneHalf * 8 + j];
            aQ1[j]     = qp[32 + laneHalf * 8 + j];
            aQ1[j + 8] = qp[48 + laneHalf * 8 + j];
        }
    }

    float m_i[8], l_i[8];
    v8f acc[4] = {};
    #pragma unroll
    for (int i = 0; i < 8; ++i) { m_i[i] = -INFINITY; l_i[i] = 0.0f; }

    for (int kc = 0; kc < Tt; kc += 32) {
        // ---- kick off async V-chunk copy (32 rows x 128B = 8 b128 per lane) ----
        #pragma unroll
        for (int jj = 0; jj < 8; ++jj) {
            int task = lane + 32 * jj;        // 0..255
            int vr   = task >> 3;             // 0..31
            int vc8  = (task & 7) * 8;        // half index, 16B chunks
            const _Float16* src = qkvh + ((size_t)(b * Tt + kc + vr)) * rowStride
                                  + headOff + 128 + vc8;
            async_copy_b128(src, &Vs[vr][vc8]);
        }

        // ---- scores: two 16x16 tiles over 32 keys, contraction hd=64 (2 WMMAs each) ----
        v8f s0 = {}, s1 = {};
        #pragma unroll
        for (int hh = 0; hh < 2; ++hh) {
            v16h bK0, bK1;
            const _Float16* kp0 = qkvh + ((size_t)(b * Tt + kc + laneMod)) * rowStride
                                  + headOff + 64 + hh * 32;
            const _Float16* kp1 = kp0 + 16 * rowStride;
            #pragma unroll
            for (int j = 0; j < 16; ++j) { int kk = (j < 8) ? laneHalf * 8 + j : 8 + laneHalf * 8 + j; bK0[j] = kp0[kk]; bK1[j] = kp1[kk]; }
            v16h aQ = hh ? aQ1 : aQ0;
            s0 = WMMA_F16(aQ, bK0, s0);
            s1 = WMMA_F16(aQ, bK1, s1);
        }

        // ---- mask*scale, online softmax over all keys ----
        float p0[8], p1[8], corr[8];
        #pragma unroll
        for (int i = 0; i < 8; ++i) {
            int q   = qBase + laneHalf * 8 + i;
            int k0i = kc + laneMod;
            int k1i = k0i + 16;
            float v0 = (k0i <= q) ? s0[i] * scale : 0.0f;
            float v1 = (k1i <= q) ? s1[i] * scale : 0.0f;
            float cm = fmaxf(v0, v1);
            cm = fmaxf(cm, __shfl_xor(cm, 1, 32));
            cm = fmaxf(cm, __shfl_xor(cm, 2, 32));
            cm = fmaxf(cm, __shfl_xor(cm, 4, 32));
            cm = fmaxf(cm, __shfl_xor(cm, 8, 32));
            float nm = fmaxf(m_i[i], cm);
            corr[i]  = __expf(m_i[i] - nm);
            p0[i]    = __expf(v0 - nm);
            p1[i]    = __expf(v1 - nm);
            float rs = p0[i] + p1[i];
            rs += __shfl_xor(rs, 1, 32);
            rs += __shfl_xor(rs, 2, 32);
            rs += __shfl_xor(rs, 4, 32);
            rs += __shfl_xor(rs, 8, 32);
            l_i[i] = l_i[i] * corr[i] + rs;
            m_i[i] = nm;
        }
        #pragma unroll
        for (int t = 0; t < 4; ++t)
            #pragma unroll
            for (int i = 0; i < 8; ++i) acc[t][i] *= corr[i];

        // ---- P -> LDS -> A-fragment layout ----
        __syncthreads();
        #pragma unroll
        for (int i = 0; i < 8; ++i) {
            Pt[laneHalf * 8 + i][laneMod]      = (_Float16)p0[i];
            Pt[laneHalf * 8 + i][16 + laneMod] = (_Float16)p1[i];
        }
        __syncthreads();
        v16h aP;
        #pragma unroll
        for (int j = 0; j < 8; ++j) {
            aP[j]     = Pt[laneMod][laneHalf * 8 + j];
            aP[j + 8] = Pt[laneMod][16 + laneHalf * 8 + j];
        }

        // ---- V chunk has landed in LDS by now (DMA overlapped the work above) ----
        wait_async0();
        __syncthreads();

        // ---- O += P @ V (4 hd-tiles of 16, K=32), B-fragments from LDS ----
        #pragma unroll
        for (int t = 0; t < 4; ++t) {
            v16h bV;
            #pragma unroll
            for (int j = 0; j < 16; ++j) bV[j] = Vs[(j < 8) ? laneHalf * 8 + j : 8 + laneHalf * 8 + j][t * 16 + laneMod];
            acc[t] = WMMA_F16(aP, bV, acc[t]);
        }
    }

    // ---- finalize: divide by rowsum, write f16 [B,T,E] at col h*64 ----
    #pragma unroll
    for (int t = 0; t < 4; ++t) {
        int col = h * 64 + t * 16 + laneMod;
        #pragma unroll
        for (int i = 0; i < 8; ++i) {
            int q = qBase + laneHalf * 8 + i;
            o[(size_t)(b * Tt + q) * Ee + col] = (_Float16)(acc[t][i] / l_i[i]);
        }
    }
}

// ---------------- LayerNorm(a + r) * g + beta, f16 in/out, f32 math ----------------
__global__ __launch_bounds__(256)
void ln_residual_kernel(const _Float16* __restrict__ a, const _Float16* __restrict__ r,
                        const float* __restrict__ g, const float* __restrict__ be,
                        _Float16* __restrict__ out)
{
    __shared__ float red[8];
    const int row = blockIdx.x;
    const int tid = threadIdx.x;
    const _Float16* pa = a + (size_t)row * Ee;
    const _Float16* pr = r + (size_t)row * Ee;
    v4h av = *(const v4h*)(pa + tid * 4);
    v4h rv = *(const v4h*)(pr + tid * 4);
    float v[4];
    float s = 0.0f;
    #pragma unroll
    for (int j = 0; j < 4; ++j) { v[j] = (float)av[j] + (float)rv[j]; s += v[j]; }
    for (int m = 16; m >= 1; m >>= 1) s += __shfl_xor(s, m, 32);
    if ((tid & 31) == 0) red[tid >> 5] = s;
    __syncthreads();
    float tot = 0.0f;
    #pragma unroll
    for (int w = 0; w < 8; ++w) tot += red[w];
    float mean = tot * (1.0f / Ee);
    float sq = 0.0f;
    #pragma unroll
    for (int j = 0; j < 4; ++j) { float d = v[j] - mean; sq += d * d; }
    for (int m = 16; m >= 1; m >>= 1) sq += __shfl_xor(sq, m, 32);
    __syncthreads();
    if ((tid & 31) == 0) red[tid >> 5] = sq;
    __syncthreads();
    float var = 0.0f;
    #pragma unroll
    for (int w = 0; w < 8; ++w) var += red[w];
    var *= (1.0f / Ee);
    float rstd = rsqrtf(var + 1e-5f);
    v4h ov;
    #pragma unroll
    for (int j = 0; j < 4; ++j) {
        int c = tid * 4 + j;
        ov[j] = (_Float16)((v[j] - mean) * rstd * g[c] + be[c]);
    }
    *(v4h*)(out + (size_t)row * Ee + tid * 4) = ov;
}

// ---------------- embedding gather (f32 table -> f16 activations) ----------------
__global__ __launch_bounds__(256)
void embed_kernel(const int* __restrict__ tokens, const float* __restrict__ emb,
                  _Float16* __restrict__ x)
{
    int row = blockIdx.x;
    int tok = tokens[row];
    float4 v = ((const float4*)(emb + (size_t)tok * Ee))[threadIdx.x];
    v4h o = { (_Float16)v.x, (_Float16)v.y, (_Float16)v.z, (_Float16)v.w };
    *(v4h*)(x + (size_t)row * Ee + threadIdx.x * 4) = o;
}

// ---------------- f32 -> f16 convert (vec4), for one-shot weight conversion ----------------
__global__ __launch_bounds__(256)
void f32_to_f16_kernel(const float* __restrict__ in, _Float16* __restrict__ out, int n4)
{
    int i = blockIdx.x * 256 + threadIdx.x;
    if (i < n4) {
        float4 v = ((const float4*)in)[i];
        v4h o = { (_Float16)v.x, (_Float16)v.y, (_Float16)v.z, (_Float16)v.w };
        *(v4h*)(out + (size_t)i * 4) = o;
    }
}

// ---------------- loss ----------------
__global__ void zero_loss_kernel(float* loss)
{
    if (threadIdx.x == 0 && blockIdx.x == 0) *loss = 0.0f;
}

__global__ __launch_bounds__(128)
void loss_kernel(const float* __restrict__ logits, const int* __restrict__ targets,
                 float* __restrict__ loss)
{
    __shared__ float red[4];
    int row = blockIdx.x;
    int t = threadIdx.x;
    float v = logits[(size_t)row * Vv + t];
    float m = v;
    for (int s = 16; s >= 1; s >>= 1) m = fmaxf(m, __shfl_xor(m, s, 32));
    if ((t & 31) == 0) red[t >> 5] = m;
    __syncthreads();
    m = fmaxf(fmaxf(red[0], red[1]), fmaxf(red[2], red[3]));
    float e = __expf(v - m);
    float se = e;
    for (int s = 16; s >= 1; s >>= 1) se += __shfl_xor(se, s, 32);
    __syncthreads();
    if ((t & 31) == 0) red[t >> 5] = se;
    __syncthreads();
    se = red[0] + red[1] + red[2] + red[3];
    float lse = m + __logf(se);
    if (t == 0) {
        int tg = targets[row];
        float nll = lse - logits[(size_t)row * Vv + tg];
        atomicAdd(loss, nll * (1.0f / (float)NTOK));
    }
}

// ---------------- host orchestration ----------------
extern "C" void kernel_launch(void* const* d_in, const int* in_sizes, int n_in,
                              void* d_out, int out_size, void* d_ws, size_t ws_size,
                              hipStream_t stream)
{
    (void)in_sizes; (void)n_in; (void)out_size; (void)ws_size;

    const int*   tokens    = (const int*)d_in[0];
    const int*   targets   = (const int*)d_in[1];
    const float* token_emb = (const float*)d_in[2];
    const float* Wqkv      = (const float*)d_in[3];
    const float* bqkv      = (const float*)d_in[4];
    const float* aw1       = (const float*)d_in[5];
    const float* ab1       = (const float*)d_in[6];
    const float* aw2       = (const float*)d_in[7];
    const float* ab2       = (const float*)d_in[8];
    const float* ln1_g     = (const float*)d_in[9];
    const float* ln1_b     = (const float*)d_in[10];
    const float* fw1       = (const float*)d_in[11];
    const float* fb1       = (const float*)d_in[12];
    const float* fw2       = (const float*)d_in[13];
    const float* fb2       = (const float*)d_in[14];
    const float* ln2_g     = (const float*)d_in[15];
    const float* ln2_b     = (const float*)d_in[16];
    const float* head_w    = (const float*)d_in[17];
    const float* head_b    = (const float*)d_in[18];

    // ---- workspace carve-out (~175 MB), all f16 except logits ----
    char* ws = (char*)d_ws;
    _Float16* WqkvH = (_Float16*)ws; ws += (size_t)Ll * Ee * E3 * 2;
    _Float16* aw1H  = (_Float16*)ws; ws += (size_t)Ll * Ee * Ee * 2;
    _Float16* aw2H  = (_Float16*)ws; ws += (size_t)Ll * Ee * Ee * 2;
    _Float16* fw1H  = (_Float16*)ws; ws += (size_t)Ll * Ee * Ee * 2;
    _Float16* fw2H  = (_Float16*)ws; ws += (size_t)Ll * Ee * Ee * 2;
    _Float16* headH = (_Float16*)ws; ws += (size_t)Ee * Vv * 2;
    _Float16* x     = (_Float16*)ws; ws += (size_t)NTOK * Ee * 2;
    _Float16* x1    = (_Float16*)ws; ws += (size_t)NTOK * Ee * 2;
    _Float16* t1    = (_Float16*)ws; ws += (size_t)NTOK * Ee * 2;
    _Float16* t2    = (_Float16*)ws; ws += (size_t)NTOK * Ee * 2;
    _Float16* qkvh  = (_Float16*)ws; ws += (size_t)NTOK * E3 * 2;

    float* logits = (float*)d_out;
    float* lossp  = logits + (size_t)NTOK * Vv;

    // ---- one-shot f32->f16 weight conversion (streamed once; GEMMs then read half the bytes) ----
    {
        int nq = Ll * Ee * E3 / 4;  // 6,291,456
        int ne = Ll * Ee * Ee / 4;  // 2,097,152
        int nh = Ee * Vv / 4;       // 32,768
        f32_to_f16_kernel<<<nq / 256, 256, 0, stream>>>(Wqkv, WqkvH, nq);
        f32_to_f16_kernel<<<ne / 256, 256, 0, stream>>>(aw1, aw1H, ne);
        f32_to_f16_kernel<<<ne / 256, 256, 0, stream>>>(aw2, aw2H, ne);
        f32_to_f16_kernel<<<ne / 256, 256, 0, stream>>>(fw1, fw1H, ne);
        f32_to_f16_kernel<<<ne / 256, 256, 0, stream>>>(fw2, fw2H, ne);
        f32_to_f16_kernel<<<nh / 256, 256, 0, stream>>>(head_w, headH, nh);
    }

    embed_kernel<<<NTOK, 256, 0, stream>>>(tokens, token_emb, x);

    const float scale = 0.125f;  // 1/sqrt(64)

    for (int l = 0; l < Ll; ++l) {
        // qkv (f16) = x @ Wqkv[l] + bqkv[l]
        gemm_f16_kernel<1><<<dim3(E3 / BN, NTOK / BM), 256, 0, stream>>>(
            x, WqkvH + (size_t)l * Ee * E3, bqkv + (size_t)l * E3, qkvh, NTOK, E3, Ee, 0);
        // attention -> t1 [NTOK, E] f16
        attn_kernel<<<dim3(Tt / 16, Hh, Bb), 32, 0, stream>>>(qkvh, t1, scale);
        // attention's internal FFN
        gemm_f16_kernel<1><<<dim3(Ee / BN, NTOK / BM), 256, 0, stream>>>(
            t1, aw1H + (size_t)l * Ee * Ee, ab1 + (size_t)l * Ee, t2, NTOK, Ee, Ee, 1);
        gemm_f16_kernel<1><<<dim3(Ee / BN, NTOK / BM), 256, 0, stream>>>(
            t2, aw2H + (size_t)l * Ee * Ee, ab2 + (size_t)l * Ee, t1, NTOK, Ee, Ee, 0);
        // x1 = LN(t1 + x)
        ln_residual_kernel<<<NTOK, 256, 0, stream>>>(
            t1, x, ln1_g + (size_t)l * Ee, ln1_b + (size_t)l * Ee, x1);
        // block FFN
        gemm_f16_kernel<1><<<dim3(Ee / BN, NTOK / BM), 256, 0, stream>>>(
            x1, fw1H + (size_t)l * Ee * Ee, fb1 + (size_t)l * Ee, t1, NTOK, Ee, Ee, 1);
        gemm_f16_kernel<1><<<dim3(Ee / BN, NTOK / BM), 256, 0, stream>>>(
            t1, fw2H + (size_t)l * Ee * Ee, fb2 + (size_t)l * Ee, t2, NTOK, Ee, Ee, 0);
        // x = LN(x1 + t2)
        ln_residual_kernel<<<NTOK, 256, 0, stream>>>(
            t2, x1, ln2_g + (size_t)l * Ee, ln2_b + (size_t)l * Ee, x);
    }

    // logits (f32, straight to d_out) = x @ head_w + head_b   (N = 128 = one BN tile)
    gemm_f16_kernel<0><<<dim3(Vv / BN, NTOK / BM), 256, 0, stream>>>(
        x, headH, head_b, logits, NTOK, Vv, Ee, 0);

    zero_loss_kernel<<<1, 32, 0, stream>>>(lossp);
    loss_kernel<<<NTOK, 128, 0, stream>>>(logits, targets, lossp);
}
